// CustomRNN_29686813950555
// MI455X (gfx1250) — matrix operand order repaired and checked by reading, not compiled, not run on hardware
//
#include <hip/hip_runtime.h>

// Problem dims (fixed by the reference)
#define BB 64
#define TT 1024
#define II 256
#define HH 512
#define MM (BB * TT)   // 65536 rows for the big GEMMs

typedef __attribute__((ext_vector_type(16))) __bf16 v16bf;
typedef __attribute__((ext_vector_type(8)))  float  v8f;
typedef __attribute__((ext_vector_type(4)))  unsigned int v4u;
typedef __attribute__((ext_vector_type(8)))  int v8i;
typedef __attribute__((ext_vector_type(4)))  int v4i;

union ABFrag { uint4 u[2]; v16bf v; };

// ---------------------------------------------------------------------------
// WMMA helper: D = A(16x32 bf16) * B(32x16 bf16) + C(16x16 f32)
// ---------------------------------------------------------------------------
__device__ __forceinline__ v8f wmma_bf16(v16bf a, v16bf b, v8f c) {
  return __builtin_amdgcn_wmma_f32_16x16x32_bf16(
      /*neg_a=*/false, a, /*neg_b=*/false, b,
      /*c_mod=*/(short)0, c, /*reuse_a=*/false, /*reuse_b=*/false);
}

// Inline, branch-free tanh: clamp then (e-1)/(e+1) with e = exp(2x).
__device__ __forceinline__ float fast_tanh(float x) {
  float xc = fminf(fmaxf(x, -15.f), 15.f);
  float e  = __expf(2.f * xc);
  return (e - 1.f) / (e + 1.f);
}

// A fragment (16x32 bf16), rows striped per ISA layout.
__device__ __forceinline__ v16bf load_frag_a(const __bf16* __restrict__ tile,
                                             int stride, int lane, int k0) {
  const int r  = lane & 15;
  const int kb = k0 + ((lane & 16) ? 8 : 0);
  const __bf16* p = tile + (size_t)r * stride + kb;
  ABFrag f;
  f.u[0] = *reinterpret_cast<const uint4*>(p);
  f.u[1] = *reinterpret_cast<const uint4*>(p + 16);
  return f.v;
}

// B fragment (32x16 bf16) from an N-major (transposed) weight matrix.
__device__ __forceinline__ v16bf load_frag_b(const __bf16* __restrict__ Wt,
                                             int K, int n0, int lane, int k0) {
  const int n  = n0 + (lane & 15);
  const int kb = k0 + ((lane & 16) ? 8 : 0);
  const __bf16* p = Wt + (size_t)n * K + kb;
  ABFrag f;
  f.u[0] = *reinterpret_cast<const uint4*>(p);
  f.u[1] = *reinterpret_cast<const uint4*>(p + 16);
  return f.v;
}

// ---------------------------------------------------------------------------
// Elementwise f32 -> bf16 conversion (x is read exactly once -> NT load)
// ---------------------------------------------------------------------------
__global__ void cvt_f32_to_bf16_kernel(const float* __restrict__ src,
                                       __bf16* __restrict__ dst, int n) {
  for (int i = blockIdx.x * blockDim.x + threadIdx.x; i < n;
       i += gridDim.x * blockDim.x)
    dst[i] = (__bf16)__builtin_nontemporal_load(&src[i]);
}

// Transpose + convert: W[k][n] f32 -> Wt[n][k] bf16 (done once, tiny)
__global__ void transpose_cvt_kernel(const float* __restrict__ W,
                                     __bf16* __restrict__ Wt, int K, int N) {
  int total = K * N;
  for (int i = blockIdx.x * blockDim.x + threadIdx.x; i < total;
       i += gridDim.x * blockDim.x) {
    int k = i / N;
    int n = i - k * N;
    Wt[(size_t)n * K + k] = (__bf16)W[i];
  }
}

// ---------------------------------------------------------------------------
// Projection GEMM: Y[M][512] = X[M][K] (bf16) @ Wt[512][K]^T (bf16), f32 out.
// ---------------------------------------------------------------------------
__global__ __launch_bounds__(256) void proj_gemm_kernel(
    const __bf16* __restrict__ X, const __bf16* __restrict__ Wt,
    float* __restrict__ Y, int K) {
  const int mtile = blockIdx.x * 16;
  const int wave  = threadIdx.x >> 5;
  const int lane  = threadIdx.x & 31;
  const int nbase = wave * 64;

  const __bf16* Atile = X + (size_t)mtile * K;

  v8f c[4];
#pragma unroll
  for (int t = 0; t < 4; ++t) c[t] = (v8f){0.f, 0.f, 0.f, 0.f, 0.f, 0.f, 0.f, 0.f};

  for (int k0 = 0; k0 < K; k0 += 32) {
    v16bf a = load_frag_a(Atile, K, lane, k0);
#pragma unroll
    for (int t = 0; t < 4; ++t) {
      v16bf b = load_frag_b(Wt, K, nbase + t * 16, lane, k0);
      c[t] = wmma_bf16(a, b, c[t]);
    }
  }

  const int moff = (lane & 16) ? 8 : 0;
  const int n    = lane & 15;
#pragma unroll
  for (int t = 0; t < 4; ++t) {
#pragma unroll
    for (int r = 0; r < 8; ++r) {
      int row = mtile + moff + r;
      int col = nbase + t * 16 + n;
      Y[(size_t)row * HH + col] = c[t][r];
    }
  }
}

// ---------------------------------------------------------------------------
// TDM: async-load one xp tile (16 rows x 512 f32, row stride T*H) into LDS.
// D# packing per CDNA5 ISA 8.3/8.4. Issued by one wave; tracked by TENSORcnt.
// 6-arg builtin form (amdgpu-toolchain / clang-23).
// ---------------------------------------------------------------------------
__device__ __forceinline__ void tdm_load_xp_tile(const float* gsrc,
                                                 unsigned int lds_off) {
  const unsigned long long ga = (unsigned long long)(uintptr_t)gsrc;
  // Group 0: count=1 | lds_addr | global_addr[56:0] | type=2
  v4u g0 = {1u,                                   // count = 1 valid descriptor
            lds_off,                              // LDS byte address
            (unsigned int)ga,                     // global_addr[31:0]
            (unsigned int)((ga >> 32) & 0x01FFFFFFu) | 0x80000000u};
  // Group 1: data_size=4B (code 2); tensor_dim0=512; tensor_dim1=65536;
  //          tile_dim0=512; tile_dim1=16; tensor_dim0_stride=T*H=524288.
  v8i g1 = {(int)(2u << 16),                      // data_size
            (int)(512u << 16),                    // tensor_dim0[15:0] @ bits63:48
            0,                                    // dim0 hi=0, dim1 lo (65536&0xffff)=0
            (int)(1u | (512u << 16)),             // dim1 hi=1 | tile_dim0=512
            16,                                   // tile_dim1=16, tile_dim2=0
            (int)(TT * HH),                       // dim0_stride[31:0]
            0, 0};
  v4i gz4 = {0, 0, 0, 0};                         // groups 2/3: 2-D tensor
  v8i gz8 = {0, 0, 0, 0, 0, 0, 0, 0};
  __builtin_amdgcn_tensor_load_to_lds(g0, g1, gz4, gz4, gz8, 0);
}

// ---------------------------------------------------------------------------
// Recurrent scan: h_t = tanh(xp[:,t,:] + h_{t-1} @ W_hh + b)
// Grid = 4 blocks (one per 16-row batch tile), 1024 threads = 32 waves.
//  * W_hh column slab register-resident per wave (16 B-frags, loaded once).
//  * h in static LDS (bf16, double buffered).
//  * xp tiles DMA'd by the Tensor Data Mover into dynamic LDS, double
//    buffered, one step ahead; synced with s_wait_tensorcnt.
// ---------------------------------------------------------------------------
__global__ __launch_bounds__(1024, 1) void rnn_scan_kernel(
    const float* __restrict__ xp,     // [B*T][H], row = b*T + t
    const __bf16* __restrict__ Whht,  // [H][H] bf16, N-major
    const float* __restrict__ bias,   // [H]
    __bf16* __restrict__ out_bf,      // [B*T][H] bf16 or nullptr
    float* __restrict__ out_f,        // [B*T][H] f32 or nullptr
    float* __restrict__ hn)           // [B][H] final hidden state
{
  __shared__ __align__(16) __bf16 hbuf[2][16][HH + 8];  // padded stride
  extern __shared__ float xpbuf[];                      // [2][16][HH] f32 (64 KB)

  const int b0   = blockIdx.x * 16;
  const int wave = threadIdx.x >> 5;     // 0..31
  const int lane = threadIdx.x & 31;
  const int n0   = wave * 16;

  // Dynamic LDS region begins right after static allocations.
  const unsigned int xpbuf_lds = __builtin_amdgcn_groupstaticsize();
  const unsigned int tile_bytes = 16u * HH * 4u;        // 32 KB per buffer

  // Preload this wave's entire W_hh column slab into registers.
  v16bf breg[16];
#pragma unroll
  for (int kk = 0; kk < 16; ++kk)
    breg[kk] = load_frag_b(Whht, HH, n0, lane, kk * 32);

  // h_0 = 0
  for (int i = threadIdx.x; i < 16 * (HH + 8); i += blockDim.x)
    (&hbuf[0][0][0])[i] = (__bf16)0.0f;

  // Prime the pipeline: DMA xp tile for t=0 into buffer 0.
  if (wave == 0) {
    tdm_load_xp_tile(xp + (size_t)b0 * TT * HH, xpbuf_lds);
    __builtin_amdgcn_s_wait_tensorcnt(0);
  }
  __syncthreads();

  const int moff = (lane & 16) ? 8 : 0;
  const int n    = lane & 15;
  const float bias_n = bias[n0 + n];

  // Per-lane base indices into out ([b*T+t][H] layout); step adds t*H.
  int xpi[8];
#pragma unroll
  for (int r = 0; r < 8; ++r)
    xpi[r] = (b0 + moff + r) * TT * HH + n0 + n;   // < 2^31, fits int

  for (int t = 0; t < TT; ++t) {
    const int cur = t & 1;
    const int nxt = cur ^ 1;
    const int toff = t * HH;

    // Kick off the DMA for the next timestep's xp tile (disjoint buffer).
    if (wave == 0 && t + 1 < TT)
      tdm_load_xp_tile(xp + (size_t)b0 * TT * HH + (t + 1) * HH,
                       xpbuf_lds + (unsigned int)nxt * tile_bytes);

    v8f c = (v8f){0.f, 0.f, 0.f, 0.f, 0.f, 0.f, 0.f, 0.f};
#pragma unroll
    for (int kk = 0; kk < 16; ++kk) {
      // A fragment from LDS (ds_load_b128)
      const int r  = lane & 15;
      const int kb = kk * 32 + ((lane & 16) ? 8 : 0);
      ABFrag fa;
      fa.u[0] = *reinterpret_cast<const uint4*>(&hbuf[cur][r][kb]);
      fa.u[1] = *reinterpret_cast<const uint4*>(&hbuf[cur][r][kb + 16]);
      c = wmma_bf16(fa.v, breg[kk], c);
    }

#pragma unroll
    for (int r = 0; r < 8; ++r) {
      const int mr = moff + r;
      const size_t idx = (size_t)(xpi[r] + toff);
      const float xpv = xpbuf[(cur * 16 + mr) * HH + n0 + n];  // DMA'd tile
      float v = fast_tanh(c[r] + xpv + bias_n);
      hbuf[nxt][mr][n0 + n] = (__bf16)v;
      if (out_bf) out_bf[idx] = (__bf16)v;
      if (out_f)  __builtin_nontemporal_store(v, &out_f[idx]);  // never re-read
      if (t == TT - 1) hn[(size_t)(b0 + mr) * HH + n0 + n] = v;
    }

    // Ensure next tile has landed before anyone crosses into step t+1.
    if (wave == 0 && t + 1 < TT)
      __builtin_amdgcn_s_wait_tensorcnt(0);
    __syncthreads();
  }
}

// ---------------------------------------------------------------------------
// Launch
// ---------------------------------------------------------------------------
extern "C" void kernel_launch(void* const* d_in, const int* in_sizes, int n_in,
                              void* d_out, int out_size, void* d_ws, size_t ws_size,
                              hipStream_t stream) {
  const float* x     = (const float*)d_in[0];   // [B,T,I]
  const float* W_ih0 = (const float*)d_in[1];   // [I,H]
  const float* W_hh0 = (const float*)d_in[2];   // [H,H]
  const float* b0    = (const float*)d_in[3];   // [H]
  const float* W_ih1 = (const float*)d_in[4];   // [H,H]
  const float* W_hh1 = (const float*)d_in[5];   // [H,H]
  const float* b1    = (const float*)d_in[6];   // [H]

  float* out1 = (float*)d_out;                      // [B,T,H]
  float* hn   = out1 + (size_t)BB * TT * HH;        // [2,B,H]

  // Workspace carve-up
  char* ws = (char*)d_ws;
  __bf16* wih0t  = (__bf16*)ws;                ws += (size_t)HH * II * 2;  // [512][256]
  __bf16* whh0t  = (__bf16*)ws;                ws += (size_t)HH * HH * 2;  // [512][512]
  __bf16* wih1t  = (__bf16*)ws;                ws += (size_t)HH * HH * 2;
  __bf16* whh1t  = (__bf16*)ws;                ws += (size_t)HH * HH * 2;
  __bf16* xbf    = (__bf16*)ws;                ws += (size_t)MM * II * 2;  // 32 MB
  float*  xp     = (float*)ws;                 ws += (size_t)MM * HH * 4;  // 128 MB (reused)
  __bf16* out0bf = (__bf16*)ws;                                            // 64 MB

  const size_t scan_dynlds = 2u * 16u * HH * 4u;   // 64 KB xp double buffer

  // 1) Weight transpose+convert (f32 [K][N] -> bf16 [N][K])
  transpose_cvt_kernel<<<256, 256, 0, stream>>>(W_ih0, wih0t, II, HH);
  transpose_cvt_kernel<<<256, 256, 0, stream>>>(W_hh0, whh0t, HH, HH);
  transpose_cvt_kernel<<<256, 256, 0, stream>>>(W_ih1, wih1t, HH, HH);
  transpose_cvt_kernel<<<256, 256, 0, stream>>>(W_hh1, whh1t, HH, HH);

  // 2) x -> bf16
  cvt_f32_to_bf16_kernel<<<4096, 256, 0, stream>>>(x, xbf, MM * II);

  // 3) Layer 0 input projection: xp = x_bf @ W_ih0 (bias folded into scan)
  proj_gemm_kernel<<<MM / 16, 256, 0, stream>>>(xbf, wih0t, xp, II);

  // 4) Layer 0 scan -> out0 (bf16 for next GEMM) + hn[0]
  rnn_scan_kernel<<<BB / 16, 1024, scan_dynlds, stream>>>(
      xp, whh0t, b0, out0bf, nullptr, hn);

  // 5) Layer 1 input projection: xp = out0_bf @ W_ih1
  proj_gemm_kernel<<<MM / 16, 256, 0, stream>>>(out0bf, wih1t, xp, HH);

  // 6) Layer 1 scan -> out1 (f32, final output) + hn[1]
  rnn_scan_kernel<<<BB / 16, 1024, scan_dynlds, stream>>>(
      xp, whh1t, b1, nullptr, out1, hn + (size_t)BB * HH);
}
